// VolSelfAttention_47923245089233
// MI455X (gfx1250) — matrix operand order repaired and checked by reading, not compiled, not run on hardware
//
#include <hip/hip_runtime.h>
#include <hip/hip_bf16.h>
#include <math.h>

// ---------------------------------------------------------------------------
// Fused windowed-attention + channel-attention (VolSelfAttention) for gfx1250
//  B=4096 windows, N=32 tokens (4x8), C=192, NH=6 heads, HD=32
//  All large GEMMs on v_wmma_f32_16x16x32_f16 (wave32 WMMA).
// ---------------------------------------------------------------------------

typedef _Float16 v16h __attribute__((ext_vector_type(16)));
typedef float    v8f  __attribute__((ext_vector_type(8)));

#define C_    192
#define NH_   6
#define HD_   32
#define N_    32
#define TC_   576          // 3*C
#define NWAVE 8            // 256 threads / wave32

// workspace byte offsets
#define OFF_QKVW   0u       // 576*192 halfs  = 221184 B
#define OFF_PROJW  221184u  // 192*192 halfs  =  73728 B
#define OFF_QKVCW  294912u  // 576*192 halfs  = 221184 B
#define OFF_PROJCW 516096u  // 192*192 halfs  =  73728 B
#define OFF_RPB    589824u  // 6*32*32 f32    =  24576 B
#define OFF_DECAY  614400u  // 6*32*32 f32    =  24576 B
#define OFF_GATE   638976u  // 32 f32
#define WS_NEED    639104u

// ---------------- WMMA fragment helpers (A/B: contiguous k = 16*(lane/16)+e) ----------------

static __device__ inline v8f wmma16(v16h a, v16h b, v8f c) {
  // D = A(16x32 f16) x B(32x16 f16) + C(16x16 f32)
  return __builtin_amdgcn_wmma_f32_16x16x32_f16(false, a, false, b, (short)0, c,
                                                false, false);
}

// A 16x32 f16, row-major source base[row*ld + k].
// lane 0-15: row=lane, K=0..15; lane 16-31: row=lane-16, K=16..31
static __device__ inline v16h load_a_frag(const _Float16* base, int ld, int lane) {
  const int r  = lane & 15;
  const int kb = (lane & 16) ? 16 : 0;
  const _Float16* p = base + r * ld + kb;
  v16h a;
#pragma unroll
  for (int e = 0; e < 16; ++e) a[e] = p[e];
  return a;
}

// B 32x16 f16 where element (k,n) lives at base[n*ld + k]  (n-major storage)
// lane 0-15: col=lane, K=0..15 ; lane 16-31: col=lane-16, K=16..31
static __device__ inline v16h load_b_nk(const _Float16* base, int ld, int lane) {
  const int n  = lane & 15;
  const int kb = (lane & 16) ? 16 : 0;
  const _Float16* p = base + n * ld + kb;
  v16h b;
#pragma unroll
  for (int e = 0; e < 16; ++e) b[e] = p[e];
  return b;
}

// B 32x16 f16 where element (k,n) lives at base[k*ld + n]  (k-major storage)
static __device__ inline v16h load_b_kn(const _Float16* base, int ld, int lane) {
  const int n  = lane & 15;
  const int kb = (lane & 16) ? 16 : 0;
  v16h b;
#pragma unroll
  for (int e = 0; e < 16; ++e) b[e] = base[(kb + e) * ld + n];
  return b;
}

// D 16x16 f32: VGPR j holds row (j + (lane>=16 ? 8:0)), col = lane&15

// ---------------------------- prep kernels ---------------------------------

__global__ void prep_convert_f16(const float* __restrict__ qkv_w,
                                 const float* __restrict__ proj_w,
                                 const float* __restrict__ qkvC_w,
                                 const float* __restrict__ projC_w,
                                 char* __restrict__ ws) {
  _Float16* h_qkvw  = (_Float16*)(ws + OFF_QKVW);
  _Float16* h_projw = (_Float16*)(ws + OFF_PROJW);
  _Float16* h_qkvcw = (_Float16*)(ws + OFF_QKVCW);
  _Float16* h_projcw= (_Float16*)(ws + OFF_PROJCW);
  int i = blockIdx.x * 256 + threadIdx.x;              // 0 .. 294911
  if (i < 110592)        h_qkvw[i]            = (_Float16)qkv_w[i];
  else if (i < 147456)   h_projw[i - 110592]  = (_Float16)proj_w[i - 110592];
  else if (i < 258048)   h_qkvcw[i - 147456]  = (_Float16)qkvC_w[i - 147456];
  else if (i < 294912)   h_projcw[i - 258048] = (_Float16)projC_w[i - 258048];
}

__global__ void prep_tables(const float* __restrict__ rpb_table,
                            const float* __restrict__ gs_w1,
                            const float* __restrict__ gs_b1,
                            const float* __restrict__ bn_g,
                            const float* __restrict__ bn_b,
                            const float* __restrict__ bn_m,
                            const float* __restrict__ bn_v,
                            const float* __restrict__ gs_w2,
                            const float* __restrict__ gs_b2,
                            char* __restrict__ ws) {
  float* f_rpb   = (float*)(ws + OFF_RPB);
  float* f_decay = (float*)(ws + OFF_DECAY);
  float* f_gate  = (float*)(ws + OFF_GATE);
  const int tid = threadIdx.x;

  // expanded relative-position bias: (NH, N, N)
  for (int e = tid; e < NH_ * N_ * N_; e += 256) {
    int h = e / (N_ * N_);
    int nm = e % (N_ * N_);
    int n = nm >> 5, m = nm & 31;
    int in_ = n >> 3, jn = n & 7;     // 4x8 grid
    int im = m >> 3, jm = m & 7;
    int idx = (in_ - im + 3) * 15 + (jn - jm + 7);
    f_rpb[e] = rpb_table[idx * NH_ + h];
  }
  // decay mask: |c-d| * log(1 - 2^(-2-4h/NH))
  for (int e = tid; e < NH_ * HD_ * HD_; e += 256) {
    int h = e / (HD_ * HD_);
    int cd = e % (HD_ * HD_);
    int c = cd >> 5, d = cd & 31;
    float dv = logf(1.0f - exp2f(-2.0f - 4.0f * (float)h / (float)NH_));
    f_decay[e] = fabsf((float)(c - d)) * dv;
  }
  // gating vector: pooled == 1/N exactly (softmax rows sum to 1)
  __shared__ float sy[32];
  if (tid < 32) {
    float y = 0.f;
    for (int h = 0; h < NH_; ++h)
      y += gs_w1[((tid * NH_ + h) * 3 + 1) * 3 + 1];
    y = y * (1.0f / (float)N_) + gs_b1[tid];
    y = (y - bn_m[tid]) * rsqrtf(bn_v[tid] + 1e-5f) * bn_g[tid] + bn_b[tid];
    sy[tid] = y;
  }
  __syncthreads();
  if (tid < 32) {
    float y = gs_b2[tid];
    for (int i = 0; i < 32; ++i)
      y += gs_w2[((tid * 32 + i) * 3 + 1) * 3 + 1] * sy[i];
    f_gate[tid] = y;
  }
}

// ------------------------------ main kernel --------------------------------
// one block (8 wave32) per window; ~144 KB dynamic LDS

__global__ __launch_bounds__(256, 1)
void fused_attn_kernel(const float* __restrict__ x,
                       const float* __restrict__ qkv_b,
                       const float* __restrict__ proj_b,
                       const float* __restrict__ temperature,
                       const float* __restrict__ dw_w,
                       const float* __restrict__ projC_b,
                       const char* __restrict__ ws,
                       float* __restrict__ out) {
  extern __shared__ char smem[];
  _Float16* s_x16 = (_Float16*)(smem);                 // 32x192 f16 : 12288
  float*    s_out = (float*)(smem + 12288);            // 32x192 f32 : 24576
  char*     scr   = smem + 36864;                      // 110592-byte scratch
  // phase A aliases
  _Float16* s_qkv   = (_Float16*)(scr);                // 32x576 f16 : 36864
  float*    s_attn  = (float*)(scr + 36864);           // 6x32x32 f32: 24576
  _Float16* s_attnh = (_Float16*)(scr + 61440);        // 6x32x32 f16: 12288
  // phase B aliases
  float*    s_qkvc  = (float*)(scr);                   // 576x32 f32 : 73728
  _Float16* s_qc    = (_Float16*)(scr + 73728);        // 192x32 f16 : 12288
  _Float16* s_kc    = (_Float16*)(scr + 86016);        // 192x32 f16 : 12288
  _Float16* s_vc    = (_Float16*)(scr + 98304);        // 192x32 f16 : 12288
  float*    s_attnC = (float*)(scr);                   // 6x32x32 f32: 24576
  _Float16* s_attnCh= (_Float16*)(scr + 24576);        // 6x32x32 f16: 12288
  _Float16* s_x2    = (_Float16*)(scr + 36864);        // 32x192 f16 : 12288

  const _Float16* h_qkvw  = (const _Float16*)(ws + OFF_QKVW);
  const _Float16* h_projw = (const _Float16*)(ws + OFF_PROJW);
  const _Float16* h_qkvcw = (const _Float16*)(ws + OFF_QKVCW);
  const _Float16* h_projcw= (const _Float16*)(ws + OFF_PROJCW);
  const float* f_rpb   = (const float*)(ws + OFF_RPB);
  const float* f_decay = (const float*)(ws + OFF_DECAY);
  const float* f_gate  = (const float*)(ws + OFF_GATE);

  const int tid  = threadIdx.x;
  const int wave = tid >> 5;
  const int lane = tid & 31;
  const int b    = blockIdx.x;
  const int dcol  = lane & 15;
  const int drow0 = (lane & 16) ? 8 : 0;

  // ---- load window x (32x192) -> f16 LDS --------------------------------
  for (int e = tid; e < N_ * C_; e += 256)
    s_x16[e] = (_Float16)x[(size_t)b * (N_ * C_) + e];
  __syncthreads();

  // ---- step 1: qkv = x @ W_qkv^T + b  (32x576) --------------------------
  for (int job = wave; job < 2 * 36; job += NWAVE) {
    const int m0 = (job & 1) * 16;
    const int o0 = (job >> 1) * 16;
    v8f acc = {};
#pragma unroll
    for (int kt = 0; kt < 6; ++kt) {
      v16h a = load_a_frag(s_x16 + m0 * C_ + kt * 32, C_, lane);
      v16h bb = load_b_nk(h_qkvw + o0 * C_ + kt * 32, C_, lane);
      acc = wmma16(a, bb, acc);
    }
    const float bias = qkv_b[o0 + dcol];
#pragma unroll
    for (int j = 0; j < 8; ++j)
      s_qkv[(m0 + drow0 + j) * TC_ + o0 + dcol] = (_Float16)(acc[j] + bias);
  }
  __syncthreads();

  // ---- step 2: attn logits q.k^T per head (6 x 32x32) -------------------
  for (int job = wave; job < NH_ * 4; job += NWAVE) {
    const int h  = job >> 2;
    const int mt = (job >> 1) & 1, nt = job & 1;
    v16h a  = load_a_frag(s_qkv + (mt * 16) * TC_ + h * HD_, TC_, lane);
    v16h bb = load_b_nk(s_qkv + (nt * 16) * TC_ + C_ + h * HD_, TC_, lane);
    v8f acc = {};
    acc = wmma16(a, bb, acc);
#pragma unroll
    for (int j = 0; j < 8; ++j)
      s_attn[h * 1024 + (mt * 16 + drow0 + j) * 32 + nt * 16 + dcol] = acc[j];
  }
  __syncthreads();

  // ---- step 3: softmax rows (scale + rpb), write f16 --------------------
  if (true) {
    const float scale = 0.17677669529663687f;   // HD^-0.5
    for (int t = tid; t < NH_ * N_; t += 256) {
      const int h = t >> 5, r = t & 31;
      float v[32], mx = -1e30f;
#pragma unroll
      for (int i = 0; i < 32; ++i) {
        v[i] = s_attn[h * 1024 + r * 32 + i] * scale + f_rpb[h * 1024 + r * 32 + i];
        mx = fmaxf(mx, v[i]);
      }
      float sum = 0.f;
#pragma unroll
      for (int i = 0; i < 32; ++i) { v[i] = __expf(v[i] - mx); sum += v[i]; }
      const float inv = 1.0f / sum;
#pragma unroll
      for (int i = 0; i < 32; ++i)
        s_attnh[h * 1024 + r * 32 + i] = (_Float16)(v[i] * inv);
    }
  }
  __syncthreads();

  // ---- step 4: h1 = attn @ v  -> overwrite q region of s_qkv ------------
  for (int job = wave; job < NH_ * 4; job += NWAVE) {
    const int h  = job >> 2;
    const int mt = (job >> 1) & 1, nt = job & 1;
    v16h a  = load_a_frag(s_attnh + h * 1024 + (mt * 16) * 32, 32, lane);
    v16h bb = load_b_kn(s_qkv + 2 * C_ + h * HD_ + nt * 16, TC_, lane);
    v8f acc = {};
    acc = wmma16(a, bb, acc);
#pragma unroll
    for (int j = 0; j < 8; ++j)
      s_qkv[(mt * 16 + drow0 + j) * TC_ + h * HD_ + nt * 16 + dcol] = (_Float16)acc[j];
  }
  __syncthreads();

  // ---- step 5: x1 = h1 @ W_proj^T + b  -> s_out (f32) -------------------
  for (int job = wave; job < 2 * 12; job += NWAVE) {
    const int m0 = (job & 1) * 16;
    const int o0 = (job >> 1) * 16;
    v8f acc = {};
#pragma unroll
    for (int kt = 0; kt < 6; ++kt) {
      v16h a  = load_a_frag(s_qkv + m0 * TC_ + kt * 32, TC_, lane);
      v16h bb = load_b_nk(h_projw + o0 * C_ + kt * 32, C_, lane);
      acc = wmma16(a, bb, acc);
    }
    const float bias = proj_b[o0 + dcol];
#pragma unroll
    for (int j = 0; j < 8; ++j)
      s_out[(m0 + drow0 + j) * C_ + o0 + dcol] = acc[j] + bias;
  }
  __syncthreads();

  // ---- step 6: channel 1x1 conv qkvc = x @ W_qkvC^T  (channel-major) ----
  for (int job = wave; job < 2 * 36; job += NWAVE) {
    const int m0 = (job & 1) * 16;
    const int o0 = (job >> 1) * 16;
    v8f acc = {};
#pragma unroll
    for (int kt = 0; kt < 6; ++kt) {
      v16h a  = load_a_frag(s_x16 + m0 * C_ + kt * 32, C_, lane);
      v16h bb = load_b_nk(h_qkvcw + o0 * C_ + kt * 32, C_, lane);
      acc = wmma16(a, bb, acc);
    }
#pragma unroll
    for (int j = 0; j < 8; ++j)
      s_qkvc[(o0 + dcol) * 32 + m0 + drow0 + j] = acc[j];   // [ch][token]
  }
  __syncthreads();

  // ---- step 7: 3x3 depthwise over 4x8 window, split into qc/kc/vc f16 ---
  for (int idx = tid; idx < TC_ * N_; idx += 256) {
    const int ch = idx >> 5, p = idx & 31;
    const int i = p >> 3, j = p & 7;
    float acc = 0.f;
#pragma unroll
    for (int di = -1; di <= 1; ++di) {
#pragma unroll
      for (int dj = -1; dj <= 1; ++dj) {
        const int ii = i + di, jj = j + dj;
        if (ii >= 0 && ii < 4 && jj >= 0 && jj < 8)
          acc += dw_w[ch * 9 + (di + 1) * 3 + (dj + 1)] *
                 s_qkvc[ch * 32 + ii * 8 + jj];
      }
    }
    if (ch < C_)            s_qc[ch * 32 + p]          = (_Float16)acc;
    else if (ch < 2 * C_)   s_kc[(ch - C_) * 32 + p]   = (_Float16)acc;
    else                    s_vc[(ch - 2 * C_) * 32 + p] = (_Float16)acc;
  }
  __syncthreads();

  // ---- step 8: l2norm over tokens for qc and kc -------------------------
  for (int t = tid; t < 2 * C_; t += 256) {
    _Float16* ptr = (t < C_) ? (s_qc + t * 32) : (s_kc + (t - C_) * 32);
    float ss = 0.f;
#pragma unroll
    for (int i = 0; i < 32; ++i) { float f = (float)ptr[i]; ss += f * f; }
    const float inv = 1.0f / fmaxf(sqrtf(ss), 1e-12f);
#pragma unroll
    for (int i = 0; i < 32; ++i) ptr[i] = (_Float16)((float)ptr[i] * inv);
  }
  __syncthreads();

  // ---- step 9: channel attention logits qc @ kc^T (contract over n) -----
  for (int job = wave; job < NH_ * 4; job += NWAVE) {
    const int h  = job >> 2;
    const int ct = (job >> 1) & 1, dt = job & 1;
    v16h a  = load_a_frag(s_qc + (h * HD_ + ct * 16) * 32, 32, lane);
    v16h bb = load_b_nk(s_kc + (h * HD_ + dt * 16) * 32, 32, lane);
    v8f acc = {};
    acc = wmma16(a, bb, acc);
#pragma unroll
    for (int j = 0; j < 8; ++j)
      s_attnC[h * 1024 + (ct * 16 + drow0 + j) * 32 + dt * 16 + dcol] = acc[j];
  }
  __syncthreads();

  // ---- step 10: softmax over d with temperature + decay -----------------
  for (int t = tid; t < NH_ * HD_; t += 256) {
    const int h = t >> 5, r = t & 31;
    const float temp = temperature[h];
    float v[32], mx = -1e30f;
#pragma unroll
    for (int i = 0; i < 32; ++i) {
      v[i] = s_attnC[h * 1024 + r * 32 + i] * temp + f_decay[h * 1024 + r * 32 + i];
      mx = fmaxf(mx, v[i]);
    }
    float sum = 0.f;
#pragma unroll
    for (int i = 0; i < 32; ++i) { v[i] = __expf(v[i] - mx); sum += v[i]; }
    const float inv = 1.0f / sum;
#pragma unroll
    for (int i = 0; i < 32; ++i)
      s_attnCh[h * 1024 + r * 32 + i] = (_Float16)(v[i] * inv);
  }
  __syncthreads();

  // ---- step 11: x2h = attnC @ vc, stored token-major f16 ----------------
  for (int job = wave; job < NH_ * 4; job += NWAVE) {
    const int h  = job >> 2;
    const int ct = (job >> 1) & 1, nt = job & 1;
    v16h a  = load_a_frag(s_attnCh + h * 1024 + (ct * 16) * 32, 32, lane);
    v16h bb = load_b_kn(s_vc + (h * HD_) * 32 + nt * 16, 32, lane);
    v8f acc = {};
    acc = wmma16(a, bb, acc);
#pragma unroll
    for (int j = 0; j < 8; ++j)   // rows = channel dim, cols = tokens
      s_x2[(nt * 16 + dcol) * C_ + h * HD_ + ct * 16 + drow0 + j] = (_Float16)acc[j];
  }
  __syncthreads();

  // ---- step 12: x2 = projC(x2h) + b ; combine with gate into s_out ------
  for (int job = wave; job < 2 * 12; job += NWAVE) {
    const int m0 = (job & 1) * 16;
    const int o0 = (job >> 1) * 16;
    v8f acc = {};
#pragma unroll
    for (int kt = 0; kt < 6; ++kt) {
      v16h a  = load_a_frag(s_x2 + m0 * C_ + kt * 32, C_, lane);
      v16h bb = load_b_nk(h_projcw + o0 * C_ + kt * 32, C_, lane);
      acc = wmma16(a, bb, acc);
    }
    const float bias = projC_b[o0 + dcol];
#pragma unroll
    for (int j = 0; j < 8; ++j) {
      const int tok = m0 + drow0 + j;
      const float x2v = acc[j] + bias;
      s_out[tok * C_ + o0 + dcol] += x2v * (1.0f + f_gate[tok]);
    }
  }
  __syncthreads();

  // ---- step 13: write result -------------------------------------------
  for (int e = tid; e < N_ * C_; e += 256)
    out[(size_t)b * (N_ * C_) + e] = s_out[e];
}

// ------------------------------- launcher ----------------------------------

extern "C" void kernel_launch(void* const* d_in, const int* in_sizes, int n_in,
                              void* d_out, int out_size, void* d_ws, size_t ws_size,
                              hipStream_t stream) {
  (void)in_sizes; (void)n_in; (void)out_size; (void)ws_size;
  const float* x         = (const float*)d_in[0];
  const float* qkv_w     = (const float*)d_in[1];
  const float* qkv_b     = (const float*)d_in[2];
  const float* proj_w    = (const float*)d_in[3];
  const float* proj_b    = (const float*)d_in[4];
  const float* rpb_table = (const float*)d_in[5];
  const float* temperature=(const float*)d_in[6];
  const float* qkvC_w    = (const float*)d_in[7];
  const float* dw_w      = (const float*)d_in[8];
  const float* projC_w   = (const float*)d_in[9];
  const float* projC_b   = (const float*)d_in[10];
  const float* gs_w1     = (const float*)d_in[11];
  const float* gs_b1     = (const float*)d_in[12];
  const float* bn_g      = (const float*)d_in[13];
  const float* bn_b      = (const float*)d_in[14];
  const float* bn_m      = (const float*)d_in[15];
  const float* bn_v      = (const float*)d_in[16];
  const float* gs_w2     = (const float*)d_in[17];
  const float* gs_b2     = (const float*)d_in[18];
  char* ws = (char*)d_ws;
  float* out = (float*)d_out;

  prep_convert_f16<<<1152, 256, 0, stream>>>(qkv_w, proj_w, qkvC_w, projC_w, ws);
  prep_tables<<<1, 256, 0, stream>>>(rpb_table, gs_w1, gs_b1, bn_g, bn_b,
                                     bn_m, bn_v, gs_w2, gs_b2, ws);

  const int smem_bytes = 147456;  // 144 KB (<= 320 KB WGP LDS)
  hipFuncSetAttribute((const void*)fused_attn_kernel,
                      hipFuncAttributeMaxDynamicSharedMemorySize, smem_bytes);
  fused_attn_kernel<<<4096, 256, smem_bytes, stream>>>(
      x, qkv_b, proj_b, temperature, dw_w, projC_b, ws, out);
}
